// Spatial3DFeatureFusion_77180562309812
// MI455X (gfx1250) — matrix operand order repaired and checked by reading, not compiled, not run on hardware
//
#include <hip/hip_runtime.h>
#include <hip/hip_bf16.h>
#include <math.h>

typedef __attribute__((ext_vector_type(16))) _Float16 v16h;
typedef __attribute__((ext_vector_type(8)))  float    v8f;

#define NLIG 512
#define NPOC 1024

// ---------------------------------------------------------------------------
// Prep: w3bar[c] = mean_k Wd3[k][c]  (Wd3 is 128x64), w3bar[64] = mean(bd3)
// ---------------------------------------------------------------------------
__global__ void prep_w3bar_kernel(const float* __restrict__ Wd3,
                                  const float* __restrict__ bd3,
                                  float* __restrict__ out) {
  int c = threadIdx.x;
  if (c < 64) {
    float s = 0.f;
    for (int k = 0; k < 128; k++) s += Wd3[k * 64 + c];
    out[c] = s * (1.f / 128.f);
  } else if (c == 64) {
    float s = 0.f;
    for (int k = 0; k < 128; k++) s += bd3[k];
    out[64] = s * (1.f / 128.f);
  }
}

// ---------------------------------------------------------------------------
// Generic linear: Y[r][t] = sum_c X[r][c] * W[t][c] + b[t]
// one block per row, blockDim.x == N, K <= 128
// ---------------------------------------------------------------------------
__global__ void linear_kernel(const float* __restrict__ X,
                              const float* __restrict__ W,
                              const float* __restrict__ b,
                              float* __restrict__ Y, int K) {
  __shared__ float sx[128];
  int r = blockIdx.x, t = threadIdx.x, N = blockDim.x;
  for (int c = t; c < K; c += N) sx[c] = X[(size_t)r * K + c];
  __syncthreads();
  float acc = b[t];
  const float* w = W + (size_t)t * K;
  for (int c = 0; c < K; c++) acc += sx[c] * w[c];
  Y[(size_t)r * N + t] = acc;
}

// ---------------------------------------------------------------------------
// Hot kernel: m[i][j] for all 512x1024 pairs via f16 WMMA.
//   e1 (K=32) built per lane from scalar dist; e2 = relu(e1 @ Wd2^T + bd2)
//   via 4x v_wmma_f32_16x16x32_f16 per 16-pair tile; m = relu(e2).w3bar + b3bar
// B tiles are staged ONCE per block in LDS (shared by all 8 waves) and each
// tile is pulled into VGPRs immediately before its WMMA, keeping only one
// B tile + one accumulator tile live (no scratch spills).
// Grid: 512 blocks x 256 threads (8 waves/block, 4096 waves total).
// ---------------------------------------------------------------------------
__global__ void __launch_bounds__(256)
pairwise_m_kernel(const float* __restrict__ lig_xyz,
                  const float* __restrict__ poc_xyz,
                  const float* __restrict__ Wd1,
                  const float* __restrict__ bd1,
                  const float* __restrict__ Wd2,
                  const float* __restrict__ bd2,
                  const float* __restrict__ w3b,
                  float* __restrict__ m) {
  __shared__ float s_w1[32], s_b1[32], s_bd2[64], s_w3[65];
  __shared__ v16h  sB[4][32];      // B operand tiles, WMMA lane layout
  int t = threadIdx.x;
  if (t < 32) { s_w1[t] = Wd1[t]; s_b1[t] = bd1[t]; }
  if (t < 64) s_bd2[t] = bd2[t];
  if (t < 65) s_w3[t] = w3b[t];
  // Build B = Wd2^T (32x64) as four 32x16 f16 tiles in WMMA layout:
  // lane holds N = lane&15, element e <-> K = 8*(lane>>4) + (e&7) + 16*(e>>3).
  if (t < 128) {
    int bt = t >> 5, bl = t & 31;
    int bg = bl >> 4, bc = bl & 15;
    v16h tmp;
#pragma unroll
    for (int e = 0; e < 16; e++)
      tmp[e] = (_Float16)Wd2[(16 * bt + bc) * 32 + 8 * bg + (e & 7) + 16 * (e >> 3)];
    sB[bt][bl] = tmp;
  }
  __syncthreads();

  int lane = t & 31;
  int wave = (blockIdx.x << 3) + (t >> 5);
  int i    = wave >> 3;          // ligand index 0..511
  int sub  = wave & 7;           // j-tile group
  int g    = lane >> 4;          // half-wave 0/1
  int col  = lane & 15;          // pair row within tile == epilogue N column

  float lx = lig_xyz[i * 3 + 0];
  float ly = lig_xyz[i * 3 + 1];
  float lz = lig_xyz[i * 3 + 2];

  // Hoisted per-lane epilogue constants (this lane's 4 output columns).
  float bd2v[4], w3v[4];
#pragma unroll
  for (int q = 0; q < 4; q++) {
    bd2v[q] = s_bd2[16 * q + col];
    w3v[q]  = s_w3[16 * q + col];
  }
  float b3bar = s_w3[64];

  for (int tt = 0; tt < 8; tt++) {
    int jt = sub * 8 + tt;
    int j  = jt * 16 + col;
    float px = poc_xyz[j * 3 + 0];
    float py = poc_xyz[j * 3 + 1];
    float pz = poc_xyz[j * 3 + 2];
    float dx = lx - px, dy = ly - py, dz = lz - pz;
    float dist = sqrtf(dx * dx + dy * dy + dz * dz);

    // A operand: e1 row for pair M=col. 16-bit A 16x32 layout:
    // elem e<8 -> K = 8g+e ; elem e>=8 -> K = 16+8g+(e-8)
    v16h A;
#pragma unroll
    for (int e = 0; e < 8; e++) {
      int k1 = 8 * g + e;
      int k2 = 16 + 8 * g + e;
      A[e]     = (_Float16)fmaxf(dist * s_w1[k1] + s_b1[k1], 0.f);
      A[e + 8] = (_Float16)fmaxf(dist * s_w1[k2] + s_b1[k2], 0.f);
    }

    float c[8];
#pragma unroll
    for (int r = 0; r < 8; r++) c[r] = 0.f;

    // Process one 16-wide output tile at a time: only one B tile and one
    // accumulator tile live across each WMMA.
#pragma unroll
    for (int q = 0; q < 4; q++) {
      v16h Bt = sB[q][lane];
      v8f z = {};
      v8f d = __builtin_amdgcn_wmma_f32_16x16x32_f16(false, A, false, Bt,
                                                     (short)0, z, false, false);
#pragma unroll
      for (int r = 0; r < 8; r++)
        c[r] += fmaxf(d[r] + bd2v[q], 0.f) * w3v[q];
    }

    // Reduce over the 16 N-columns (lanes within each half-wave).
#pragma unroll
    for (int r = 0; r < 8; r++) {
      c[r] += __shfl_xor(c[r], 1, 32);
      c[r] += __shfl_xor(c[r], 2, 32);
      c[r] += __shfl_xor(c[r], 4, 32);
      c[r] += __shfl_xor(c[r], 8, 32);
    }
    if (col == 0) {
      int jbase = jt * 16 + 8 * g;   // C/D VGPR r holds pair row r + 8g
#pragma unroll
      for (int r = 0; r < 8; r++)
        m[(size_t)i * NPOC + jbase + r] = c[r] + b3bar;
    }
  }
}

// ---------------------------------------------------------------------------
// out[r][t] = softmax_j(-m_row(r)) . B[j][t]
// element j of row r lives at m[r*rstride + j*jstride]; blockDim = 128
// ---------------------------------------------------------------------------
__global__ void softmax_agg_kernel(const float* __restrict__ m,
                                   const float* __restrict__ B,
                                   float* __restrict__ out,
                                   int NJ, int jstride, int rstride) {
  __shared__ float sw[1024];
  __shared__ float red[128];
  int r = blockIdx.x, t = threadIdx.x;
  const float* mr = m + (size_t)r * rstride;
  float mx = -3.4e38f;
  for (int j = t; j < NJ; j += 128) {
    float s = -mr[(size_t)j * jstride];
    sw[j] = s;
    mx = fmaxf(mx, s);
  }
  red[t] = mx; __syncthreads();
  for (int off = 64; off > 0; off >>= 1) {
    if (t < off) red[t] = fmaxf(red[t], red[t + off]);
    __syncthreads();
  }
  mx = red[0]; __syncthreads();
  float z = 0.f;
  for (int j = t; j < NJ; j += 128) {
    float e = expf(sw[j] - mx);
    sw[j] = e; z += e;
  }
  red[t] = z; __syncthreads();
  for (int off = 64; off > 0; off >>= 1) {
    if (t < off) red[t] += red[t + off];
    __syncthreads();
  }
  float invZ = 1.f / red[0];
  float acc = 0.f;
  for (int j = 0; j < NJ; j++) acc += sw[j] * B[(size_t)j * 128 + t];
  out[(size_t)r * 128 + t] = acc * invZ;
}

// ---------------------------------------------------------------------------
// gate = sigmoid([x, a] @ Wg^T + bg);  out = gate*x + (1-gate)*a.  blockDim=128
// ---------------------------------------------------------------------------
__global__ void gate_enh_kernel(const float* __restrict__ X,
                                const float* __restrict__ A,
                                const float* __restrict__ Wg,
                                const float* __restrict__ bg,
                                float* __restrict__ out) {
  __shared__ float sx[128], sa[128];
  int r = blockIdx.x, t = threadIdx.x;
  sx[t] = X[(size_t)r * 128 + t];
  sa[t] = A[(size_t)r * 128 + t];
  __syncthreads();
  float acc = bg[t];
  const float* w = Wg + (size_t)t * 256;
  for (int c = 0; c < 128; c++) acc += sx[c] * w[c];
  for (int c = 0; c < 128; c++) acc += sa[c] * w[128 + c];
  float gte = 1.f / (1.f + expf(-acc));
  out[(size_t)r * 128 + t] = gte * sx[t] + (1.f - gte) * sa[t];
}

// ---------------------------------------------------------------------------
// Attention for one (query, head): qkv rows are [q(128)|k(128)|v(128)].
// grid = (NQ, 8), blockDim = 256. Output att[q][h*16+d] (head-major columns).
// ---------------------------------------------------------------------------
__global__ void attn_kernel(const float* __restrict__ qsrc,
                            const float* __restrict__ kvsrc,
                            float* __restrict__ att, int NK) {
  __shared__ float sq[16];
  __shared__ float ss[1024];
  __shared__ float red[256];
  int qi = blockIdx.x, h = blockIdx.y, t = threadIdx.x;
  if (t < 16) sq[t] = qsrc[(size_t)qi * 384 + h * 16 + t];
  __syncthreads();
  float mx = -3.4e38f;
  for (int j = t; j < NK; j += 256) {
    const float* kp = kvsrc + (size_t)j * 384 + 128 + h * 16;
    float s = 0.f;
#pragma unroll
    for (int d = 0; d < 16; d++) s += sq[d] * kp[d];
    s *= 0.25f;                 // 1/sqrt(16)
    ss[j] = s;
    mx = fmaxf(mx, s);
  }
  red[t] = mx; __syncthreads();
  for (int off = 128; off > 0; off >>= 1) {
    if (t < off) red[t] = fmaxf(red[t], red[t + off]);
    __syncthreads();
  }
  mx = red[0]; __syncthreads();
  float z = 0.f;
  for (int j = t; j < NK; j += 256) {
    float e = expf(ss[j] - mx);
    ss[j] = e; z += e;
  }
  red[t] = z; __syncthreads();
  for (int off = 128; off > 0; off >>= 1) {
    if (t < off) red[t] += red[t + off];
    __syncthreads();
  }
  float invZ = 1.f / red[0];
  __syncthreads();
  int d = t & 15, ch = t >> 4;   // 16 chunks x 16 dims
  float acc = 0.f;
  for (int j = ch; j < NK; j += 16)
    acc += ss[j] * kvsrc[(size_t)j * 384 + 256 + h * 16 + d];
  red[t] = acc; __syncthreads();
  for (int off = 8; off > 0; off >>= 1) {
    if (ch < off) red[ch * 16 + d] += red[(ch + off) * 16 + d];
    __syncthreads();
  }
  if (ch == 0) att[(size_t)qi * 128 + h * 16 + d] = red[d] * invZ;
}

// ---------------------------------------------------------------------------
// out = LayerNorm(X + R) * g + b.  blockDim = 128 (one row).
// ---------------------------------------------------------------------------
__global__ void ln_residual_kernel(const float* __restrict__ X,
                                   const float* __restrict__ R,
                                   const float* __restrict__ g,
                                   const float* __restrict__ b,
                                   float* __restrict__ out) {
  __shared__ float red[128];
  int r = blockIdx.x, t = threadIdx.x;
  float x = X[(size_t)r * 128 + t] + R[(size_t)r * 128 + t];
  red[t] = x; __syncthreads();
  for (int off = 64; off > 0; off >>= 1) {
    if (t < off) red[t] += red[t + off];
    __syncthreads();
  }
  float mean = red[0] * (1.f / 128.f); __syncthreads();
  float dxm = x - mean;
  red[t] = dxm * dxm; __syncthreads();
  for (int off = 64; off > 0; off >>= 1) {
    if (t < off) red[t] += red[t + off];
    __syncthreads();
  }
  float var = red[0] * (1.f / 128.f);
  out[(size_t)r * 128 + t] = dxm * rsqrtf(var + 1e-5f) * g[t] + b[t];
}

// ---------------------------------------------------------------------------
extern "C" void kernel_launch(void* const* d_in, const int* in_sizes, int n_in,
                              void* d_out, int out_size, void* d_ws, size_t ws_size,
                              hipStream_t stream) {
  (void)in_sizes; (void)n_in; (void)out_size; (void)ws_size;
  const float* ligF = (const float*)d_in[0];
  const float* pocF = (const float*)d_in[1];
  const float* ligX = (const float*)d_in[2];
  const float* pocX = (const float*)d_in[3];
  const float* Wl   = (const float*)d_in[4];
  const float* bl   = (const float*)d_in[5];
  const float* Wp   = (const float*)d_in[6];
  const float* bp   = (const float*)d_in[7];
  const float* Wd1  = (const float*)d_in[8];
  const float* bd1  = (const float*)d_in[9];
  const float* Wd2  = (const float*)d_in[10];
  const float* bd2  = (const float*)d_in[11];
  const float* Wd3  = (const float*)d_in[12];
  const float* bd3  = (const float*)d_in[13];
  const float* Wgl  = (const float*)d_in[14];
  const float* bgl  = (const float*)d_in[15];
  const float* Wgp  = (const float*)d_in[16];
  const float* bgp  = (const float*)d_in[17];
  const float* Wqkv = (const float*)d_in[18];
  const float* bqkv = (const float*)d_in[19];
  const float* Wo   = (const float*)d_in[20];
  const float* bo   = (const float*)d_in[21];
  const float* g_l  = (const float*)d_in[22];
  const float* be_l = (const float*)d_in[23];
  const float* g_p  = (const float*)d_in[24];
  const float* be_p = (const float*)d_in[25];

  float* ws    = (float*)d_ws;
  float* w3b   = ws;                    // 128 (65 used)
  float* lig   = w3b   + 128;           // 512*128
  float* poc   = lig   + 65536;         // 1024*128
  float* mbuf  = poc   + 131072;        // 512*1024
  float* agg_l = mbuf  + 524288;        // 512*128
  float* agg_p = agg_l + 65536;         // 1024*128
  float* lig_e = agg_p + 131072;        // 512*128
  float* poc_e = lig_e + 65536;         // 1024*128
  float* qkv_l = poc_e + 131072;        // 512*384
  float* qkv_p = qkv_l + 196608;        // 1024*384
  float* att_l = qkv_p + 393216;        // 512*128
  float* att_p = att_l + 65536;         // 1024*128
  float* prj_l = att_p + 131072;        // 512*128
  float* prj_p = prj_l + 65536;         // 1024*128

  float* out_l = (float*)d_out;         // 512*128
  float* out_p = out_l + 65536;         // 1024*128

  // 1. w3bar / b3bar (collapses the 64->128 layer + mean into a 64-dot)
  prep_w3bar_kernel<<<dim3(1), dim3(128), 0, stream>>>(Wd3, bd3, w3b);

  // 2. input projections
  linear_kernel<<<dim3(NLIG), dim3(128), 0, stream>>>(ligF, Wl, bl, lig, 128);
  linear_kernel<<<dim3(NPOC), dim3(128), 0, stream>>>(pocF, Wp, bp, poc, 128);

  // 3. pairwise distance-MLP (WMMA hot kernel) -> m (512x1024)
  pairwise_m_kernel<<<dim3(512), dim3(256), 0, stream>>>(
      ligX, pocX, Wd1, bd1, Wd2, bd2, w3b, mbuf);

  // 4. bidirectional softmax aggregation
  softmax_agg_kernel<<<dim3(NLIG), dim3(128), 0, stream>>>(mbuf, poc, agg_l, NPOC, 1, NPOC);
  softmax_agg_kernel<<<dim3(NPOC), dim3(128), 0, stream>>>(mbuf, lig, agg_p, NLIG, NPOC, 1);

  // 5. gated enhancement
  gate_enh_kernel<<<dim3(NLIG), dim3(128), 0, stream>>>(lig, agg_l, Wgl, bgl, lig_e);
  gate_enh_kernel<<<dim3(NPOC), dim3(128), 0, stream>>>(poc, agg_p, Wgp, bgp, poc_e);

  // 6. QKV projections (one fused N=384 GEMM per side)
  linear_kernel<<<dim3(NLIG), dim3(384), 0, stream>>>(lig_e, Wqkv, bqkv, qkv_l, 128);
  linear_kernel<<<dim3(NPOC), dim3(384), 0, stream>>>(poc_e, Wqkv, bqkv, qkv_p, 128);

  // 7. cross attention both directions
  attn_kernel<<<dim3(NLIG, 8), dim3(256), 0, stream>>>(qkv_l, qkv_p, att_l, NPOC);
  attn_kernel<<<dim3(NPOC, 8), dim3(256), 0, stream>>>(qkv_p, qkv_l, att_p, NLIG);

  // 8. output projections
  linear_kernel<<<dim3(NLIG), dim3(128), 0, stream>>>(att_l, Wo, bo, prj_l, 128);
  linear_kernel<<<dim3(NPOC), dim3(128), 0, stream>>>(att_p, Wo, bo, prj_p, 128);

  // 9. residual + LayerNorm -> outputs
  ln_residual_kernel<<<dim3(NLIG), dim3(128), 0, stream>>>(lig_e, prj_l, g_l, be_l, out_l);
  ln_residual_kernel<<<dim3(NPOC), dim3(128), 0, stream>>>(poc_e, prj_p, g_p, be_p, out_p);
}
